// ResidualVQ_34076270527003
// MI455X (gfx1250) — compile-verified, matrix operand order checked by reading, not run on hardware
//
#include <hip/hip_runtime.h>
#include <hip/hip_bf16.h>
#include <math.h>

#define NUM_Q 8
#define DD    1024
#define CDIM  32
#define KCB   1024
#define BB    16
#define TLEN  4096
#define TTILE 32
#define NTILES_T (TLEN / TTILE)        /* 128 */
#define NUM_TILES (BB * NTILES_T)      /* 2048 */

#define R_STRIDE   1028                /* f32, pad 4 -> bank spread, 16B aligned rows */
#define ZE_STRIDE  36                  /* f32 */
#define EBF_STRIDE 40                  /* bf16: 80B rows, 16B aligned, bank-spread */

typedef __attribute__((ext_vector_type(16))) __bf16 v16bf;
typedef __attribute__((ext_vector_type(8)))  float  v8f;

union ABf { v16bf bf; uint4 q[2]; };

/* ---- LDS layout (bytes) ---- */
#define R_OFF     0
#define ZEP_OFF   (R_OFF + TTILE * R_STRIDE * 4)            /* 2 split-K partials  */
#define ZE_OFF    (ZEP_OFF + 2 * TTILE * ZE_STRIDE * 4)     /* biased z_e (f32)    */
#define ENC_OFF   (ZE_OFF + TTILE * ZE_STRIDE * 4)          /* normalized (bf16)   */
#define ZQ_OFF    (ENC_OFF + TTILE * EBF_STRIDE * 2)        /* gathered z_q (bf16) */
#define CANDV_OFF (ZQ_OFF + TTILE * EBF_STRIDE * 2)
#define CANDI_OFF (CANDV_OFF + 8 * TTILE * 4)
#define IDX_OFF   (CANDI_OFF + 8 * TTILE * 4)
#define LOSS_OFF  (IDX_OFF + TTILE * 4)
#define LDS_BYTES (LOSS_OFF + 8 * 4)                        /* = 152,736 B */

/* Load one lane's 16 bf16 A/B fragment (row-major 16b row) per ISA 16-bit layout:
   lanes 0-15 hold K {0..7,16..23}, lanes 16-31 hold K {8..15,24..31}. */
__device__ __forceinline__ ABf load_frag_row(const __bf16* rowbase, int sel) {
  ABf r;
  const char* p = (const char*)rowbase;
  r.q[0] = *(const uint4*)(p + sel * 16);
  r.q[1] = *(const uint4*)(p + 32 + sel * 16);
  return r;
}

__device__ __forceinline__ v8f wmma_bf16(const ABf& a, const ABf& b, v8f c) {
  return __builtin_amdgcn_wmma_f32_16x16x32_bf16(false, a.bf, false, b.bf,
                                                 (short)0, c, false, false);
}

/* ---------- prep: bf16 weight copies + L2-normalized bf16 codebook ---------- */
__global__ __launch_bounds__(256) void prep_kernel(
    const float* __restrict__ in_w, const float* __restrict__ out_w,
    const float* __restrict__ cb,
    __bf16* __restrict__ in_w_bf, __bf16* __restrict__ out_w_bf,
    __bf16* __restrict__ cbn_bf) {
  int i = blockIdx.x * 256 + threadIdx.x;              /* < NUM_Q*CDIM*DD == NUM_Q*KCB*CDIM */
  in_w_bf[i]  = (__bf16)in_w[i];
  out_w_bf[i] = (__bf16)out_w[i];
  int row = i >> 5;
  const float* rp = cb + (size_t)row * CDIM;
  float ss = 0.f;
  #pragma unroll
  for (int c = 0; c < CDIM; ++c) ss += rp[c] * rp[c];
  float inv = 1.f / fmaxf(sqrtf(ss), 1e-12f);
  cbn_bf[i] = (__bf16)(cb[i] * inv);
}

/* ---------- main: one WG = one (batch, 32-timestep) tile, 8 stages in LDS ---------- */
__global__ __launch_bounds__(256) void rvq_main_kernel(
    const float* __restrict__ x,
    const float* __restrict__ in_b,  const float* __restrict__ out_b,
    const float* __restrict__ codebook,
    const __bf16* __restrict__ in_w_bf, const __bf16* __restrict__ cbn_bf,
    const __bf16* __restrict__ out_w_bf,
    float* __restrict__ q_out, int* __restrict__ idx_out,
    float* __restrict__ loss_part) {
  extern __shared__ char smem[];
  float*  rS    = (float*)(smem + R_OFF);      /* [TTILE][R_STRIDE], r[t][d]   */
  float*  zeP   = (float*)(smem + ZEP_OFF);    /* [2][TTILE][ZE_STRIDE]        */
  float*  zeS   = (float*)(smem + ZE_OFF);     /* [TTILE][ZE_STRIDE]           */
  __bf16* encS  = (__bf16*)(smem + ENC_OFF);   /* [TTILE][EBF_STRIDE]          */
  __bf16* zqS   = (__bf16*)(smem + ZQ_OFF);    /* [TTILE][EBF_STRIDE]          */
  float*  candV = (float*)(smem + CANDV_OFF);
  int*    candI = (int*)(smem + CANDI_OFF);
  int*    idxS  = (int*)(smem + IDX_OFF);
  float*  lossS = (float*)(smem + LOSS_OFF);

  const int tid = threadIdx.x;
  const int ln  = tid & 31;
  const int wv  = tid >> 5;
  const int sel = ln >> 4;        /* K-half owned by this lane */
  const int nn  = ln & 15;        /* row (A) / column (B,C) within 16x16 tile */
  const int b   = blockIdx.y;
  const int t0  = blockIdx.x * TTILE;

  /* load x tile -> rS transposed to [t][d] (coalesced over t) */
  for (int i = tid; i < TTILE * DD; i += 256) {
    int d = i >> 5, t = i & 31;
    rS[t * R_STRIDE + d] = x[((size_t)b * DD + d) * TLEN + t0 + t];
  }
  __syncthreads();

  for (int q = 0; q < NUM_Q; ++q) {
    /* ---- Phase A: z_e[32x32] = in_w (32x1024) x r (1024x32); 2-way split-K ---- */
    {
      const __bf16* Aw = in_w_bf + (size_t)q * CDIM * DD;
      const int mt = (wv >> 1) & 1, nt = wv & 1, kh = wv >> 2;
      const int t  = nt * 16 + nn;
      const __bf16* arow = Aw + (mt * 16 + nn) * DD;
      v8f acc = {};
      #pragma unroll 4
      for (int ks = 0; ks < 16; ++ks) {
        const int kb = kh * 512 + ks * 32;
        ABf a = load_frag_row(arow + kb, sel);
        const float* rr = rS + t * R_STRIDE + kb + sel * 8;
        float tmp[16];
        *(float4*)&tmp[0]  = *(const float4*)(rr);
        *(float4*)&tmp[4]  = *(const float4*)(rr + 4);
        *(float4*)&tmp[8]  = *(const float4*)(rr + 16);
        *(float4*)&tmp[12] = *(const float4*)(rr + 20);
        ABf bm;
        #pragma unroll
        for (int j = 0; j < 16; ++j) bm.bf[j] = (__bf16)tmp[j];
        acc = wmma_bf16(a, bm, acc);
      }
      float* zp = zeP + kh * (TTILE * ZE_STRIDE);
      #pragma unroll
      for (int v = 0; v < 8; ++v)
        zp[t * ZE_STRIDE + mt * 16 + sel * 8 + v] = acc[v];  /* unique owner */
    }
    __syncthreads();

    /* ---- Phase B: fixed-order split-K reduce + bias + L2 normalize ---- */
    if (tid < TTILE) {
      const int t = tid;
      float zv[CDIM]; float ss = 0.f;
      #pragma unroll
      for (int c = 0; c < CDIM; ++c) {
        float v = zeP[t * ZE_STRIDE + c] +
                  zeP[TTILE * ZE_STRIDE + t * ZE_STRIDE + c] + in_b[q * CDIM + c];
        zv[c] = v; ss += v * v;
      }
      float inv = 1.f / fmaxf(sqrtf(ss), 1e-12f);
      #pragma unroll
      for (int c = 0; c < CDIM; ++c) {
        zeS[t * ZE_STRIDE + c] = zv[c];
        encS[t * EBF_STRIDE + c] = (__bf16)(zv[c] * inv);
      }
    }
    __syncthreads();

    /* ---- Phase C: sim = cbn x enc, streaming argmax (wave owns 128 codes) ---- */
    {
      ABf eb[2];
      #pragma unroll
      for (int nt = 0; nt < 2; ++nt)
        eb[nt] = load_frag_row(encS + (nt * 16 + nn) * EBF_STRIDE, sel);
      float bestV[2] = { -3.0e38f, -3.0e38f };
      int   bestI[2] = { 0, 0 };
      const __bf16* Cw = cbn_bf + (size_t)q * KCB * CDIM;
      #pragma unroll
      for (int mt = 0; mt < 8; ++mt) {
        const int code0 = wv * 128 + mt * 16;
        ABf a = load_frag_row(Cw + (code0 + nn) * CDIM, sel);
        #pragma unroll
        for (int nt = 0; nt < 2; ++nt) {
          v8f c = {};
          c = wmma_bf16(a, eb[nt], c);
          #pragma unroll
          for (int v = 0; v < 8; ++v) {
            float val = c[v]; int code = code0 + sel * 8 + v;
            if (val > bestV[nt] || (val == bestV[nt] && code < bestI[nt])) {
              bestV[nt] = val; bestI[nt] = code;
            }
          }
        }
      }
      #pragma unroll
      for (int nt = 0; nt < 2; ++nt) {
        float ov = __shfl_down(bestV[nt], 16, 32);
        int   oi = __shfl_down(bestI[nt], 16, 32);
        if (ov > bestV[nt] || (ov == bestV[nt] && oi < bestI[nt])) {
          bestV[nt] = ov; bestI[nt] = oi;
        }
        if (ln < 16) {
          candV[wv * TTILE + nt * 16 + ln] = bestV[nt];
          candI[wv * TTILE + nt * 16 + ln] = bestI[nt];
        }
      }
    }
    __syncthreads();
    if (tid < TTILE) {
      float bv = -3.0e38f; int bi = 0x7fffffff;
      #pragma unroll
      for (int w = 0; w < 8; ++w) {
        float v = candV[w * TTILE + tid]; int i = candI[w * TTILE + tid];
        if (v > bv || (v == bv && i < bi)) { bv = v; bi = i; }
      }
      idxS[tid] = bi;
      idx_out[((size_t)q * BB + b) * TLEN + t0 + tid] = bi;
    }
    __syncthreads();

    /* ---- Phase D: gather z_q (bf16 to LDS) + commitment/codebook loss partial ---- */
    {
      const int t = tid >> 3, g = tid & 7;
      const int code = idxS[t];
      float4 zq4 = *(const float4*)(codebook + ((size_t)q * KCB + code) * CDIM + g * 4);
      float d0 = zeS[t * ZE_STRIDE + g * 4 + 0] - zq4.x;
      float d1 = zeS[t * ZE_STRIDE + g * 4 + 1] - zq4.y;
      float d2 = zeS[t * ZE_STRIDE + g * 4 + 2] - zq4.z;
      float d3 = zeS[t * ZE_STRIDE + g * 4 + 3] - zq4.w;
      float ssd = d0 * d0 + d1 * d1 + d2 * d2 + d3 * d3;
      zqS[t * EBF_STRIDE + g * 4 + 0] = (__bf16)zq4.x;
      zqS[t * EBF_STRIDE + g * 4 + 1] = (__bf16)zq4.y;
      zqS[t * EBF_STRIDE + g * 4 + 2] = (__bf16)zq4.z;
      zqS[t * EBF_STRIDE + g * 4 + 3] = (__bf16)zq4.w;
      #pragma unroll
      for (int off = 16; off > 0; off >>= 1) ssd += __shfl_down(ssd, off, 32);
      if (ln == 0) lossS[wv] = ssd;
    }
    __syncthreads();
    if (tid == 0) {
      float s = 0.f;
      #pragma unroll
      for (int w = 0; w < 8; ++w) s += lossS[w];
      loss_part[q * NUM_TILES + b * NTILES_T + blockIdx.x] = s;
    }

    /* ---- Phase E: out = out_w (1024x32) x z_q + b_out; r -= out (ds_add_f32) ---- */
    {
      ABf zb[2];
      #pragma unroll
      for (int nt = 0; nt < 2; ++nt)
        zb[nt] = load_frag_row(zqS + (nt * 16 + nn) * EBF_STRIDE, sel);
      const __bf16* Ow = out_w_bf + (size_t)q * DD * CDIM;
      #pragma unroll
      for (int mt = 0; mt < 8; ++mt) {
        const int d0r = wv * 128 + mt * 16;
        ABf a = load_frag_row(Ow + (d0r + nn) * CDIM, sel);
        float bia[8];
        #pragma unroll
        for (int v = 0; v < 8; ++v) bia[v] = out_b[q * DD + d0r + sel * 8 + v];
        #pragma unroll
        for (int nt = 0; nt < 2; ++nt) {
          v8f c = {};
          c = wmma_bf16(a, zb[nt], c);
          const int t = nt * 16 + nn;
          #pragma unroll
          for (int v = 0; v < 8; ++v)
            atomicAdd(&rS[t * R_STRIDE + d0r + sel * 8 + v], -(c[v] + bia[v]));
        }
      }
    }
    __syncthreads();
  }

  /* quantized_out = x - residual_final */
  for (int i = tid; i < TTILE * DD; i += 256) {
    int d = i >> 5, t = i & 31;
    size_t gi = ((size_t)b * DD + d) * TLEN + t0 + t;
    q_out[gi] = x[gi] - rS[t * R_STRIDE + d];
  }
}

/* ---------- deterministic fixed-order loss reduction: loss[q] = 1.25*SSD/(32*T*B) ---------- */
__global__ __launch_bounds__(256) void loss_reduce_kernel(
    const float* __restrict__ part, float* __restrict__ loss_out) {
  __shared__ float sh[256];
  const int q = blockIdx.x;
  float s = 0.f;
  for (int i = threadIdx.x; i < NUM_TILES; i += 256) s += part[q * NUM_TILES + i];
  sh[threadIdx.x] = s;
  __syncthreads();
  for (int o = 128; o > 0; o >>= 1) {
    if (threadIdx.x < o) sh[threadIdx.x] += sh[threadIdx.x + o];
    __syncthreads();
  }
  if (threadIdx.x == 0)
    loss_out[q] = sh[0] * (1.25f / ((float)CDIM * (float)TLEN * (float)BB));
}

extern "C" void kernel_launch(void* const* d_in, const int* in_sizes, int n_in,
                              void* d_out, int out_size, void* d_ws, size_t ws_size,
                              hipStream_t stream) {
  (void)in_sizes; (void)n_in; (void)out_size; (void)ws_size;
  const float* x     = (const float*)d_in[0];
  const float* in_w  = (const float*)d_in[1];
  const float* in_b  = (const float*)d_in[2];
  const float* out_w = (const float*)d_in[3];
  const float* out_b = (const float*)d_in[4];
  const float* cb    = (const float*)d_in[5];

  float* q_out    = (float*)d_out;
  int*   idx_out  = (int*)(q_out + (size_t)BB * DD * TLEN);
  float* loss_out = q_out + (size_t)BB * DD * TLEN + (size_t)NUM_Q * BB * TLEN;

  char* ws = (char*)d_ws;
  __bf16* in_w_bf  = (__bf16*)(ws);
  __bf16* out_w_bf = (__bf16*)(ws + 524288);
  __bf16* cbn_bf   = (__bf16*)(ws + 1048576);
  float*  loss_part = (float*)(ws + 1572864);

  (void)hipFuncSetAttribute((const void*)rvq_main_kernel,
                            hipFuncAttributeMaxDynamicSharedMemorySize, LDS_BYTES);

  prep_kernel<<<dim3((NUM_Q * CDIM * DD) / 256), dim3(256), 0, stream>>>(
      in_w, out_w, cb, in_w_bf, out_w_bf, cbn_bf);

  rvq_main_kernel<<<dim3(NTILES_T, BB), dim3(256), LDS_BYTES, stream>>>(
      x, in_b, out_b, cb, in_w_bf, cbn_bf, out_w_bf, q_out, idx_out, loss_part);

  loss_reduce_kernel<<<dim3(NUM_Q), dim3(256), 0, stream>>>(loss_part, loss_out);
}
